// Attention_1803886265625
// MI455X (gfx1250) — matrix operand order repaired and checked by reading, not compiled, not run on hardware
//
#include <hip/hip_runtime.h>
#include <hip/hip_bf16.h>

// Problem constants (from the reference)
#define B_   2
#define S_   2048
#define H_   16
#define DH_  64
#define DM_  1024
#define BS_  (B_*S_)          // 4096

typedef _Float16 f16;
typedef __attribute__((ext_vector_type(8)))  _Float16 v8h;
typedef __attribute__((ext_vector_type(16))) _Float16 v16h;
typedef __attribute__((ext_vector_type(8)))  float    v8f;

#define NEG_BIG (-1.0e30f)

// ---- fragment loaders (two 16B loads -> v16h) ----
__device__ __forceinline__ v16h ldpair(const f16* lo, const f16* hi) {
  v8h a = *(const v8h*)lo;
  v8h b = *(const v8h*)hi;
  return __builtin_shufflevector(a, b, 0,1,2,3,4,5,6,7,8,9,10,11,12,13,14,15);
}
// A-frag (16x32 f16): lane holds row m, 16 contiguous k starting at kk + (lane<16?0:16) (koA = half*8)
__device__ __forceinline__ v16h ldA(const f16* row, int kk, int koA) {
  return ldpair(row + kk + 2 * koA, row + kk + 2 * koA + 8);
}
// B-frag (32x16 f16): lane holds column n, 16 contiguous k starting at (lane<16?0:16)
__device__ __forceinline__ v16h ldB(const f16* col) { return ldpair(col, col + 8); }

__device__ __forceinline__ v8f wmma_f16(v16h a, v16h b, v8f c) {
  return __builtin_amdgcn_wmma_f32_16x16x32_f16(false, a, false, b, (short)0, c, false, false);
}

// half-wave (16 lane) reductions; xor masks 1,2,4,8 stay inside each 16-lane half of the wave32
__device__ __forceinline__ float hmax16(float v) {
  v = fmaxf(v, __shfl_xor(v, 1));
  v = fmaxf(v, __shfl_xor(v, 2));
  v = fmaxf(v, __shfl_xor(v, 4));
  v = fmaxf(v, __shfl_xor(v, 8));
  return v;
}
__device__ __forceinline__ float hsum16(float v) {
  v += __shfl_xor(v, 1);
  v += __shfl_xor(v, 2);
  v += __shfl_xor(v, 4);
  v += __shfl_xor(v, 8);
  return v;
}

// ---------------- conversion kernels ----------------
__global__ void cvt_x_f16(const float* __restrict__ in, f16* __restrict__ out, int n) {
  int i = blockIdx.x * blockDim.x + threadIdx.x;
  if (i < n) out[i] = (f16)in[i];
}
// W_{Q,K,V}[h][m][d] -> Wt[h][d][m] (f16); n = H_*DH_*DM_ = 2^20
__global__ void cvt_wqkv(const float* __restrict__ in, f16* __restrict__ out) {
  int i = blockIdx.x * blockDim.x + threadIdx.x;
  int m = i & (DM_ - 1);
  int d = (i >> 10) & (DH_ - 1);
  int h = i >> 16;
  out[i] = (f16)in[((size_t)h * DM_ + m) * DH_ + d];
}
// W_O[k=h*64+d][m] -> Wot[m][k] (f16); n = DM_*DM_ = 2^20
__global__ void cvt_wo(const float* __restrict__ in, f16* __restrict__ out) {
  int i = blockIdx.x * blockDim.x + threadIdx.x;
  int k = i & (DM_ - 1);
  int m = i >> 10;
  out[i] = (f16)in[(size_t)k * DM_ + m];
}

// ---------------- QKV projection ----------------
// one wave -> 32 rows x 64 cols of {Q|K|V} for one head (B-fragments reused by both row
// subtiles: 12 b128 loads per 8 WMMAs). waves = 3*16*128 = 6144.
__global__ void qkv_kernel(const f16* __restrict__ xh,   // [BS][DM]
                           const f16* __restrict__ wt,   // [3][H][DH][DM]
                           const float* __restrict__ bQ,
                           const float* __restrict__ bK,
                           const float* __restrict__ bV,
                           f16* __restrict__ Qh,         // [B][H][S][DH] (pre-scaled by 1/8)
                           f16* __restrict__ Kh,         // [B][H][S][DH]
                           f16* __restrict__ Vt)         // [B][H][DH][S]
{
  int wave = blockIdx.x * (blockDim.x >> 5) + (threadIdx.x >> 5);
  int lane = threadIdx.x & 31;
  int rt = wave & 127;          // 128 tiles of 32 rows
  int h  = (wave >> 7) & 15;
  int w  = wave >> 11;          // 0=Q 1=K 2=V
  int r0 = rt * 32;
  int n    = lane & 15;
  int half = lane >> 4;
  int koA = half * 8;
  int koB = half * 16;

  const f16* arow0 = xh + (size_t)(r0 + n) * DM_;
  const f16* arow1 = xh + (size_t)(r0 + 16 + n) * DM_;
  v8f acc[2][4] = {};

  for (int kk = 0; kk < DM_; kk += 32) {
    v16h a0 = ldA(arow0, kk, koA);
    v16h a1 = ldA(arow1, kk, koA);
#pragma unroll
    for (int t = 0; t < 4; t++) {
      const f16* bp = wt + ((((size_t)w * H_ + h) * DH_) + t * 16 + n) * DM_ + kk + koB;
      v16h b = ldB(bp);
      acc[0][t] = wmma_f16(a0, b, acc[0][t]);
      acc[1][t] = wmma_f16(a1, b, acc[1][t]);
    }
  }

  const float* bias = (w == 0) ? bQ : ((w == 1) ? bK : bV);

#pragma unroll
  for (int u = 0; u < 2; u++) {
    int r0u = r0 + u * 16;
    int bb = r0u >> 11;           // /S_
    int s0 = r0u & (S_ - 1);
    if (w == 2) {
      // V transposed store: lane owns column d; rows j+half*8 contiguous in s -> one 16B store/tile
#pragma unroll
      for (int t = 0; t < 4; t++) {
        int d = t * 16 + n;
        float bv = bias[h * DH_ + d];
        v8h pk;
#pragma unroll
        for (int j = 0; j < 8; j++) pk[j] = (f16)(acc[u][t][j] + bv);
        size_t off = (((size_t)(bb * H_ + h) * DH_ + d) * S_) + s0 + half * 8;
        *(v8h*)(Vt + off) = pk;
      }
    } else {
      f16* dst = (w == 0) ? Qh : Kh;
      float sc = (w == 0) ? 0.125f : 1.0f;   // fold 1/sqrt(d_head) into Q
#pragma unroll
      for (int t = 0; t < 4; t++) {
        int d = t * 16 + n;
        float bv = bias[h * DH_ + d];
#pragma unroll
        for (int j = 0; j < 8; j++) {
          int s = s0 + j + half * 8;
          dst[(((size_t)(bb * H_ + h) * S_ + s) * DH_) + d] = (f16)((acc[u][t][j] + bv) * sc);
        }
      }
    }
  }
}

// ---------------- causal flash attention ----------------
// one wave -> one (b, h, 32-query tile); K/V fragments reused by both query subtiles
// (16 b128 loads per 16 WMMAs). waves = 2*16*64 = 2048.
__global__ void attn_kernel(const f16* __restrict__ Qh,
                            const f16* __restrict__ Kh,
                            const f16* __restrict__ Vt,
                            f16* __restrict__ Zh)       // [B][S][H*DH]
{
  __shared__ __align__(16) f16 pbuf[4][2][16][40];   // per-wave 2x(16x32) P tiles (padded rows)
  int wslot = threadIdx.x >> 5;
  int wave  = blockIdx.x * 4 + wslot;
  int lane  = threadIdx.x & 31;
  int qt = wave & 63;
  int h  = (wave >> 6) & 15;
  int bb = wave >> 10;
  int q0 = qt * 32;
  int n    = lane & 15;
  int half = lane >> 4;
  int koA = half * 8;
  int koB = half * 16;

  const f16* Qbase = Qh + (size_t)(bb * H_ + h) * S_ * DH_;
  const f16* Kbase = Kh + (size_t)(bb * H_ + h) * S_ * DH_;
  const f16* Vbase = Vt + (size_t)(bb * H_ + h) * DH_ * S_;

  // Q A-fragments for both query subtiles, d = 0..31 and 32..63 (loaded once)
  const f16* qrow0 = Qbase + (size_t)(q0 + n) * DH_;
  const f16* qrow1 = Qbase + (size_t)(q0 + 16 + n) * DH_;
  v16h qa[2][2];
  qa[0][0] = ldA(qrow0, 0, koA);  qa[0][1] = ldA(qrow0, 32, koA);
  qa[1][0] = ldA(qrow1, 0, koA);  qa[1][1] = ldA(qrow1, 32, koA);

  float mrun[2][8], lrun[2][8];
  v8f z[2][4] = {};
#pragma unroll
  for (int u = 0; u < 2; u++)
#pragma unroll
    for (int j = 0; j < 8; j++) { mrun[u][j] = NEG_BIG; lrun[u][j] = 0.0f; }

  int kend = q0 + 32;                       // causal: keys 0 .. q0+31
  for (int ks = 0; ks < kend; ks += 32) {
    // K fragments for this 32-key step (shared by both query subtiles)
    const f16* kp0 = Kbase + (size_t)(ks + n) * DH_;
    const f16* kp1 = Kbase + (size_t)(ks + 16 + n) * DH_;
    v16h kb00 = ldB(kp0 + koB), kb01 = ldB(kp0 + 32 + koB);
    v16h kb10 = ldB(kp1 + koB), kb11 = ldB(kp1 + 32 + koB);

    v8f s[2][2] = {};
#pragma unroll
    for (int u = 0; u < 2; u++) {
      s[u][0] = wmma_f16(qa[u][0], kb00, s[u][0]);
      s[u][0] = wmma_f16(qa[u][1], kb01, s[u][0]);
      s[u][1] = wmma_f16(qa[u][0], kb10, s[u][1]);
      s[u][1] = wmma_f16(qa[u][1], kb11, s[u][1]);
    }

#pragma unroll
    for (int u = 0; u < 2; u++) {
      int q0u = q0 + u * 16;
      // causal mask (only steps touching/after the subtile's diagonal)
      if (ks + 31 > q0u) {
#pragma unroll
        for (int j = 0; j < 8; j++) {
          int q = q0u + j + half * 8;
          if (ks + n > q)      s[u][0][j] = NEG_BIG;
          if (ks + 16 + n > q) s[u][1][j] = NEG_BIG;
        }
      }
      // online softmax update; row m = j + half*8 lives in the 16-lane half at VGPR j
#pragma unroll
      for (int j = 0; j < 8; j++) {
        float mc = hmax16(fmaxf(s[u][0][j], s[u][1][j]));
        float mn = fmaxf(mrun[u][j], mc);
        float scale = __expf(mrun[u][j] - mn);
        float p0 = __expf(s[u][0][j] - mn);
        float p1 = __expf(s[u][1][j] - mn);
        float rs = hsum16(p0 + p1);
        lrun[u][j] = lrun[u][j] * scale + rs;
        mrun[u][j] = mn;
#pragma unroll
        for (int t = 0; t < 4; t++) z[u][t][j] *= scale;
        int r = j + half * 8;
        pbuf[wslot][u][r][n]      = (f16)p0;
        pbuf[wslot][u][r][16 + n] = (f16)p1;
      }
    }

    // P (16x32) as A-fragments via LDS (same-wave; DS deps handled by dscnt waits)
    v16h pa0 = ldA(&pbuf[wslot][0][n][0], 0, koA);
    v16h pa1 = ldA(&pbuf[wslot][1][n][0], 0, koA);
    // z += P * V : 4 column tiles over d; each V fragment feeds both query subtiles
#pragma unroll
    for (int t = 0; t < 4; t++) {
      const f16* vp = Vbase + (size_t)(t * 16 + n) * S_ + ks + koB;
      v16h vb = ldB(vp);
      z[0][t] = wmma_f16(pa0, vb, z[0][t]);
      z[1][t] = wmma_f16(pa1, vb, z[1][t]);
    }
  }

  // epilogue: normalize and store z as f16 [B][S][H*DH]
#pragma unroll
  for (int u = 0; u < 2; u++) {
#pragma unroll
    for (int t = 0; t < 4; t++) {
      int d = t * 16 + n;
#pragma unroll
      for (int j = 0; j < 8; j++) {
        int q = q0 + u * 16 + j + half * 8;
        float val = z[u][t][j] / lrun[u][j];
        Zh[((size_t)(bb * S_ + q)) * DM_ + h * DH_ + d] = (f16)val;
      }
    }
  }
}

// ---------------- output projection ----------------
// out[BS][DM] = Z[BS][DM] @ Wot + b_O, 32x64 tile per wave. waves = 128*16 = 2048.
__global__ void proj_kernel(const f16* __restrict__ Zh,
                            const f16* __restrict__ Wot,  // [DM(n)][DM(k)]
                            const float* __restrict__ bO,
                            float* __restrict__ out)
{
  int wave = blockIdx.x * (blockDim.x >> 5) + (threadIdx.x >> 5);
  int lane = threadIdx.x & 31;
  int rt = wave & 127;         // 128 tiles of 32 rows
  int ct = wave >> 7;          // 0..15 (64-wide column tiles)
  int r0 = rt * 32;
  int c0 = ct * 64;
  int n    = lane & 15;
  int half = lane >> 4;
  int koA = half * 8;
  int koB = half * 16;

  const f16* arow0 = Zh + (size_t)(r0 + n) * DM_;
  const f16* arow1 = Zh + (size_t)(r0 + 16 + n) * DM_;
  v8f acc[2][4] = {};

  for (int kk = 0; kk < DM_; kk += 32) {
    v16h a0 = ldA(arow0, kk, koA);
    v16h a1 = ldA(arow1, kk, koA);
#pragma unroll
    for (int t = 0; t < 4; t++) {
      const f16* bp = Wot + (size_t)(c0 + t * 16 + n) * DM_ + kk + koB;
      v16h b = ldB(bp);
      acc[0][t] = wmma_f16(a0, b, acc[0][t]);
      acc[1][t] = wmma_f16(a1, b, acc[1][t]);
    }
  }

#pragma unroll
  for (int u = 0; u < 2; u++) {
#pragma unroll
    for (int t = 0; t < 4; t++) {
      int c = c0 + t * 16 + n;
      float bv = bO[c];
#pragma unroll
      for (int j = 0; j < 8; j++) {
        out[(size_t)(r0 + u * 16 + j + half * 8) * DM_ + c] = acc[u][t][j] + bv;
      }
    }
  }
}

// ---------------- launch ----------------
extern "C" void kernel_launch(void* const* d_in, const int* in_sizes, int n_in,
                              void* d_out, int out_size, void* d_ws, size_t ws_size,
                              hipStream_t stream) {
  (void)in_sizes; (void)n_in; (void)out_size; (void)ws_size;
  const float* x  = (const float*)d_in[0];
  const float* WQ = (const float*)d_in[1];
  const float* WK = (const float*)d_in[2];
  const float* WV = (const float*)d_in[3];
  const float* WO = (const float*)d_in[4];
  const float* bQ = (const float*)d_in[5];
  const float* bK = (const float*)d_in[6];
  const float* bV = (const float*)d_in[7];
  const float* bO = (const float*)d_in[8];
  float* out = (float*)d_out;

  // workspace carve (48 MB total, all 256B-aligned offsets)
  char* ws = (char*)d_ws;
  f16* XH  = (f16*)(ws + 0);              // 8 MB : x in f16 [BS][DM]
  f16* WT  = (f16*)(ws + (8u  << 20));    // 6 MB : [3][H][DH][DM]
  f16* WOT = (f16*)(ws + (14u << 20));    // 2 MB : [DM][DM] (transposed W_O)
  f16* QH  = (f16*)(ws + (16u << 20));    // 8 MB
  f16* KH  = (f16*)(ws + (24u << 20));    // 8 MB
  f16* VT  = (f16*)(ws + (32u << 20));    // 8 MB
  f16* ZH  = (f16*)(ws + (40u << 20));    // 8 MB

  cvt_x_f16<<<(BS_ * DM_) / 256, 256, 0, stream>>>(x, XH, BS_ * DM_);
  cvt_wqkv<<<(H_ * DH_ * DM_) / 256, 256, 0, stream>>>(WQ, WT + 0 * (H_ * DH_ * DM_));
  cvt_wqkv<<<(H_ * DH_ * DM_) / 256, 256, 0, stream>>>(WK, WT + 1 * (H_ * DH_ * DM_));
  cvt_wqkv<<<(H_ * DH_ * DM_) / 256, 256, 0, stream>>>(WV, WT + 2 * (H_ * DH_ * DM_));
  cvt_wo<<<(DM_ * DM_) / 256, 256, 0, stream>>>(WO, WOT);

  // 3 * 16 heads * 128 row-tiles = 6144 waves; 4 waves/block
  qkv_kernel<<<1536, 128, 0, stream>>>(XH, WT, bQ, bK, bV, QH, KH, VT);
  // 2 * 16 * 64 q-tiles = 2048 waves
  attn_kernel<<<512, 128, 0, stream>>>(QH, KH, VT, ZH);
  // 128 row-tiles * 16 col-tiles = 2048 waves
  proj_kernel<<<512, 128, 0, stream>>>(ZH, WOT, bO, out);
}
